// BertScoreModel_61332132987645
// MI455X (gfx1250) — compile-verified, matrix operand order checked by reading, not run on hardware
//
#include <hip/hip_runtime.h>
#include <hip/hip_bf16.h>

#define NL 4
#define Bsz 64
#define L1 256
#define L2 256
#define Dd 1024
#define LDA 40            // padded LDS row stride in halves (80B, 16B-aligned)
#define NEGBIG (-3.0e38f)

typedef _Float16 v16h __attribute__((ext_vector_type(16)));
typedef _Float16 v8h  __attribute__((ext_vector_type(8)));
typedef _Float16 v4h  __attribute__((ext_vector_type(4)));
typedef float    v8f  __attribute__((ext_vector_type(8)));

union Frag16 { v16h v; v8h half8[2]; };

// ---------------------------------------------------------------------------
// Kernel A: reciprocal L2 norms of every (l,b,i) row of reps1 and reps2.
// One wave32 per 1024-float row: 8 float4 loads/lane, shuffle reduce.
// ---------------------------------------------------------------------------
__global__ __launch_bounds__(256) void norms_kernel(
    const float* __restrict__ r1, const float* __restrict__ r2,
    float* __restrict__ invn1, float* __restrict__ invn2) {
  const int wid  = blockIdx.x * 8 + (threadIdx.x >> 5);   // global wave id == row id
  const int lane = threadIdx.x & 31;
  const int rows1 = NL * Bsz * L1;
  const float* src;
  float* dst;
  long row;
  if (wid < rows1) { src = r1; dst = invn1; row = wid; }
  else             { src = r2; dst = invn2; row = wid - rows1; }
  const float4* p = (const float4*)(src + row * (long)Dd);
  float s = 0.f;
#pragma unroll
  for (int i = 0; i < 8; ++i) {
    float4 v = p[i * 32 + lane];
    s += v.x * v.x + v.y * v.y + v.z * v.z + v.w * v.w;
  }
#pragma unroll
  for (int m = 16; m; m >>= 1) s += __shfl_xor(s, m, 32);
  if (lane == 0) dst[row] = 1.0f / sqrtf(s);
}

// ---------------------------------------------------------------------------
// Kernel B: per (l,b) fused 256x256x1024 GEMM (f16 WMMA, f32 accum) +
// cosine scaling + ragged masked row-max / col-max reductions -> s1, s2.
// WG = 256 threads = 8 waves; 2 M-passes of 128 rows; K tiled by 32.
// Wave (wm in 0..3, wn in 0..1) owns a 32x128 tile = 2x8 WMMA accumulators.
// ---------------------------------------------------------------------------
__global__ __launch_bounds__(256) void score_kernel(
    const float* __restrict__ r1, const float* __restrict__ r2,
    const int* __restrict__ len1, const int* __restrict__ len2,
    const float* __restrict__ invn1, const float* __restrict__ invn2,
    float* __restrict__ s1buf, float* __restrict__ s2buf) {
  __shared__ _Float16 lds_a[128 * LDA];   // 10.0 KB
  __shared__ _Float16 lds_b[256 * LDA];   // 20.0 KB
  __shared__ float sh_in1[256];
  __shared__ float sh_in2[256];
  __shared__ float rowmax_sh[2][128];
  __shared__ float colmax_sh[4][256];
  __shared__ float red[256];

  const int lb   = blockIdx.x;          // l*B + b
  const int b    = lb & (Bsz - 1);
  const int t    = threadIdx.x;
  const int lane = t & 31;
  const int w    = t >> 5;
  const int c    = lane & 15;           // column-in-tile / row-in-tile lane index
  const int h    = lane >> 4;           // k-half selector per ISA A/B layout
  const int wm   = w & 3;               // 32-row chunk
  const int wn   = w >> 2;              // 128-col chunk
  const int n1   = len1[b];
  const int n2   = len2[b];

  const long base = (long)lb * L1 * Dd;
  const float* Ag = r1 + base;
  const float* Bg = r2 + base;

  sh_in1[t] = invn1[lb * L1 + t];
  sh_in2[t] = invn2[lb * L2 + t];

  float rowpart = 0.f;        // threads t<128: sum over valid rows of rowmax
  float colreg  = NEGBIG;     // thread t: running max over all rows for col j=t

  for (int mp = 0; mp < 2; ++mp) {
    const int m_base = mp * 128;

    v8f acc[2][8];
#pragma unroll
    for (int mi = 0; mi < 2; ++mi)
#pragma unroll
      for (int nj = 0; nj < 8; ++nj) acc[mi][nj] = v8f{};

    for (int ks = 0; ks < 32; ++ks) {
      const int k0 = ks * 32;
      __syncthreads();   // LDS safe to overwrite
      {
        const int q  = t & 7;       // float4 within the 32-wide k slice
        const int rr = t >> 3;      // 0..31 rows per sweep
#pragma unroll
        for (int it = 0; it < 4; ++it) {
          const int r = it * 32 + rr;
          const float4 v = *(const float4*)(Ag + (long)(m_base + r) * Dd + k0 + q * 4);
          v4h hv; hv.x = (_Float16)v.x; hv.y = (_Float16)v.y;
                  hv.z = (_Float16)v.z; hv.w = (_Float16)v.w;
          *(v4h*)(lds_a + r * LDA + q * 4) = hv;
        }
#pragma unroll
        for (int it = 0; it < 8; ++it) {
          const int r = it * 32 + rr;
          const float4 v = *(const float4*)(Bg + (long)r * Dd + k0 + q * 4);
          v4h hv; hv.x = (_Float16)v.x; hv.y = (_Float16)v.y;
                  hv.z = (_Float16)v.z; hv.w = (_Float16)v.w;
          *(v4h*)(lds_b + r * LDA + q * 4) = hv;
        }
      }
      __syncthreads();

      // A fragments: lane holds row (l%16), halves [8h,8h+8) and [16+8h,..)
      Frag16 af[2];
#pragma unroll
      for (int mi = 0; mi < 2; ++mi) {
        const int row = wm * 32 + mi * 16 + c;
        af[mi].half8[0] = *(const v8h*)(lds_a + row * LDA + 8 * h);
        af[mi].half8[1] = *(const v8h*)(lds_a + row * LDA + 16 + 8 * h);
      }
#pragma unroll
      for (int nj = 0; nj < 8; ++nj) {
        const int row = wn * 128 + nj * 16 + c;   // column j of dots = row of h2
        Frag16 bf;
        bf.half8[0] = *(const v8h*)(lds_b + row * LDA + 8 * h);
        bf.half8[1] = *(const v8h*)(lds_b + row * LDA + 16 + 8 * h);
        acc[0][nj] = __builtin_amdgcn_wmma_f32_16x16x32_f16(
            false, af[0].v, false, bf.v, (short)0, acc[0][nj], false, false);
        acc[1][nj] = __builtin_amdgcn_wmma_f32_16x16x32_f16(
            false, af[1].v, false, bf.v, (short)0, acc[1][nj], false, false);
      }
    }

    // ---- epilogue: scale by inv norms, masked row/col maxes ----
    float rm[16];
    float cm[8];
#pragma unroll
    for (int k = 0; k < 16; ++k) rm[k] = NEGBIG;
#pragma unroll
    for (int k = 0; k < 8; ++k) cm[k] = NEGBIG;

#pragma unroll
    for (int mi = 0; mi < 2; ++mi) {
#pragma unroll
      for (int nj = 0; nj < 8; ++nj) {
        const int j   = wn * 128 + nj * 16 + c;
        const float sj = sh_in2[j];
        const bool jv  = (j < n2);
#pragma unroll
        for (int r = 0; r < 8; ++r) {
          // C/D layout: VGPR r = row (r + 8h) of the 16x16 tile, lane c = col
          const int il = m_base + wm * 32 + mi * 16 + 8 * h + r;  // 0..255
          const float v  = acc[mi][nj][r] * sh_in1[il] * sj;
          const float xi = (il < n1) ? v : NEGBIG;   // feeds col-max
          const float xj = jv ? v : NEGBIG;          // feeds row-max
          cm[nj]        = fmaxf(cm[nj], xi);
          rm[mi * 8 + r] = fmaxf(rm[mi * 8 + r], xj);
        }
      }
    }
    // row-max: butterfly across the 16 lanes sharing h (cols of this wave tile)
#pragma unroll
    for (int s = 1; s < 16; s <<= 1)
#pragma unroll
      for (int k = 0; k < 16; ++k)
        rm[k] = fmaxf(rm[k], __shfl_xor(rm[k], s, 32));
    if (c == 0) {
#pragma unroll
      for (int k = 0; k < 16; ++k) {
        const int mi = k >> 3, r = k & 7;
        rowmax_sh[wn][wm * 32 + mi * 16 + 8 * h + r] = rm[k];
      }
    }
    // col-max: combine the two h-halves (rows 0-7 vs 8-15 of each tile)
#pragma unroll
    for (int nj = 0; nj < 8; ++nj)
      cm[nj] = fmaxf(cm[nj], __shfl_xor(cm[nj], 16, 32));
    if (h == 0) {
#pragma unroll
      for (int nj = 0; nj < 8; ++nj)
        colmax_sh[wm][wn * 128 + nj * 16 + c] = cm[nj];
    }
    __syncthreads();

    if (t < 128) {
      const float rv = fmaxf(rowmax_sh[0][t], rowmax_sh[1][t]);
      const int ig = m_base + t;
      rowpart += (ig < n1) ? rv : 0.f;
    }
    {
      const float cv = fmaxf(fmaxf(colmax_sh[0][t], colmax_sh[1][t]),
                             fmaxf(colmax_sh[2][t], colmax_sh[3][t]));
      colreg = fmaxf(colreg, cv);
    }
    // next pass's k-loop __syncthreads() guards the shared buffers
  }

  // deterministic tree reductions (no atomics)
  red[t] = (t < 128) ? rowpart : 0.f;
  __syncthreads();
#pragma unroll
  for (int s = 128; s > 0; s >>= 1) {
    if (t < s) red[t] += red[t + s];
    __syncthreads();
  }
  const float rowsum = red[0];
  __syncthreads();
  red[t] = (t < n2) ? colreg : 0.f;
  __syncthreads();
#pragma unroll
  for (int s = 128; s > 0; s >>= 1) {
    if (t < s) red[t] += red[t + s];
    __syncthreads();
  }
  if (t == 0) {
    s1buf[lb] = red[0] / (float)n2;    // mean over valid cols of col-max
    s2buf[lb] = rowsum / (float)n1;    // mean over valid rows of row-max
  }
}

// ---------------------------------------------------------------------------
// Kernel C: F1 harmonic mean, BatchNorm over batch (biased var), linear head.
// One block, 256 threads: thread t <-> (layer t/64, batch t%64).
// ---------------------------------------------------------------------------
__global__ __launch_bounds__(256) void head_kernel(
    const float* __restrict__ s1buf, const float* __restrict__ s2buf,
    const float* __restrict__ wv, const float* __restrict__ bias,
    float* __restrict__ out) {
  __shared__ float feat[NL][Bsz];
  const int t  = threadIdx.x;
  const int li = t >> 6;
  const int bi = t & 63;
  const float s1 = s1buf[li * Bsz + bi];
  const float s2 = s2buf[li * Bsz + bi];
  const float f  = 2.f * s1 * s2 / (s1 + s2);
  feat[li][bi] = f;
  __syncthreads();
  float m = 0.f;
#pragma unroll 8
  for (int k = 0; k < Bsz; ++k) m += feat[li][k];
  m *= (1.f / Bsz);
  float v = 0.f;
#pragma unroll 8
  for (int k = 0; k < Bsz; ++k) { const float d = feat[li][k] - m; v += d * d; }
  v *= (1.f / Bsz);
  const float nf = (f - m) / sqrtf(v + 1e-8f);
  __syncthreads();
  feat[li][bi] = nf;
  __syncthreads();
  if (t < Bsz) {
    float o = bias[0];
#pragma unroll
    for (int l = 0; l < NL; ++l) o += wv[l] * feat[l][t];
    out[t] = o;   // LOGIT_SCALE == 1.0
  }
}

extern "C" void kernel_launch(void* const* d_in, const int* in_sizes, int n_in,
                              void* d_out, int out_size, void* d_ws, size_t ws_size,
                              hipStream_t stream) {
  const float* r1   = (const float*)d_in[0];   // (NL,B,L1,D) f32
  const float* r2   = (const float*)d_in[1];   // (NL,B,L2,D) f32
  const int*   len1 = (const int*)d_in[2];     // (B,) i32
  const int*   len2 = (const int*)d_in[3];     // (B,) i32
  const float* wv   = (const float*)d_in[4];   // (1,NL) f32
  const float* bias = (const float*)d_in[5];   // (1,) f32
  float* out = (float*)d_out;                  // (B,) f32

  float* ws    = (float*)d_ws;
  float* invn1 = ws;                           // NL*B*L1
  float* invn2 = invn1 + NL * Bsz * L1;        // NL*B*L2
  float* s1buf = invn2 + NL * Bsz * L2;        // NL*B
  float* s2buf = s1buf + NL * Bsz;             // NL*B

  const int total_rows = NL * Bsz * (L1 + L2);  // 131072 rows, 8 waves/block
  norms_kernel<<<total_rows / 8, 256, 0, stream>>>(r1, r2, invn1, invn2);
  score_kernel<<<NL * Bsz, 256, 0, stream>>>(r1, r2, len1, len2,
                                             invn1, invn2, s1buf, s2buf);
  head_kernel<<<1, 256, 0, stream>>>(s1buf, s2buf, wv, bias, out);
}
